// GNN2_9689446219778
// MI455X (gfx1250) — matrix operand order repaired and checked by reading, not compiled, not run on hardware
//
#include <hip/hip_runtime.h>
#include <cstdint>

typedef __attribute__((ext_vector_type(16))) __bf16 v16bf;
typedef __attribute__((ext_vector_type(8)))  float  v8f;

#define N_FEAT   657
#define N_NODES  64
#define BATCH    256
#define M_TOT    16384          // BATCH * N_NODES
#define KPAD     672            // 21 * 32
#define NPAD     704            // 44 * 16
#define KT_CHEB  21             // KPAD / 32
#define NT_CHEB  44             // NPAD / 16
#define FIN      42048          // 64 * 657
#define KT_FC1   1314           // FIN / 32
#define EPS      1e-5f

__device__ __forceinline__ float softplusf(float x) {
    return fmaxf(x, 0.f) + log1pf(expf(-fabsf(x)));
}
__device__ __forceinline__ float bn_apply(const float* __restrict__ p, int n, int i, float x) {
    return p[i] * (x - p[2*n + i]) * rsqrtf(p[3*n + i] + EPS) + p[n + i];
}
__device__ __forceinline__ v8f wmma_bf16(v16bf a, v16bf b, v8f c) {
    return __builtin_amdgcn_wmma_f32_16x16x32_bf16(false, a, false, b, (short)0, c, false, false);
}
// A-fragment swizzle: element e of lane (h,lr) covers K = e<8 ? h*8+e : 16+h*8+(e-8)
__device__ __forceinline__ int a_kk(int h, int e) {
    return (e < 8) ? (h * 8 + e) : (16 + h * 8 + (e - 8));
}

// ---------------------------------------------------------------- K0: build S
__global__ __launch_bounds__(512) void build_S(const float* __restrict__ adj,
                                               const int* __restrict__ ei,
                                               float* __restrict__ S) {
    __shared__ float wsum[512];
    __shared__ float dinv[64];
    __shared__ float Sl[4096];
    const int t = threadIdx.x;
    for (int i = t; i < 4096; i += 512) Sl[i] = 0.f;
    const int s = ei[t], d = ei[512 + t];
    float acc = 0.f;
    for (int b = 0; b < BATCH; ++b) {
        const float* p = adj + (size_t)b * 4096;
        acc += fmaxf(p[s * 64 + d], p[d * 64 + s]);
    }
    wsum[t] = acc;
    __syncthreads();
    if (t < 64) {
        float deg = 0.f;
        for (int j = 0; j < 8; ++j) deg += wsum[t * 8 + j];
        dinv[t] = deg > 0.f ? rsqrtf(deg) : 0.f;
    }
    __syncthreads();
    Sl[d * 64 + s] = -dinv[s] * wsum[t] * dinv[d];   // (d,s) pairs unique
    __syncthreads();
    for (int i = t; i < 4096; i += 512) S[i] = Sl[i];
}

// ------------------------------------------------- K1/K2: Y = scale * (S @ X)
__global__ void spmm64(const float* __restrict__ S, const float* __restrict__ X,
                       float* __restrict__ Y, float scale) {
    int idx = blockIdx.x * 256 + threadIdx.x;
    if (idx >= N_NODES * N_FEAT) return;
    int n = idx / N_FEAT, c = idx % N_FEAT;
    float acc = 0.f;
    for (int k = 0; k < 64; ++k) acc += S[n * 64 + k] * X[k * N_FEAT + c];
    Y[idx] = scale * acc;
}

// --------------------------------------- K3: cor = h1 @ W1 + h2 @ W2 (64x657)
__global__ void cor_kernel(const float* __restrict__ h1, const float* __restrict__ h2,
                           const float* __restrict__ cheb, float* __restrict__ cor) {
    int idx = blockIdx.x * 256 + threadIdx.x;
    if (idx >= N_NODES * N_FEAT) return;
    int r = idx / N_FEAT, c = idx % N_FEAT;
    const float* W1 = cheb + (size_t)N_FEAT * N_FEAT;
    const float* W2 = cheb + 2 * (size_t)N_FEAT * N_FEAT;
    float acc = 0.f;
    for (int k = 0; k < N_FEAT; ++k)
        acc += h1[r * N_FEAT + k] * W1[(size_t)k * N_FEAT + c]
             + h2[r * N_FEAT + k] * W2[(size_t)k * N_FEAT + c];
    cor[idx] = acc;
}

// --------------------------- bf16 converts into WMMA fragment-order layouts
// A-swizzle: [mtile][ktile][lane(32)][e(16)], lane = h*16 + row%16
__global__ void conv_x_swz(const float* __restrict__ x, __bf16* __restrict__ A) {
    int idx = blockIdx.x * 256 + threadIdx.x;          // over 1024*21*512
    if (idx >= (M_TOT / 16) * KT_CHEB * 512) return;
    int t = idx >> 9, lane = (idx >> 4) & 31, e = idx & 15;
    int mtile = t / KT_CHEB, ktile = t % KT_CHEB;
    int h = lane >> 4, lr = lane & 15;
    int r = mtile * 16 + lr;
    int c = ktile * 32 + a_kk(h, e);
    A[idx] = (c < N_FEAT) ? (__bf16)x[(size_t)r * N_FEAT + c] : (__bf16)0.f;
}
// B-swizzle: [ktile][ntile][lane(32)][e(16)], lane = h*16 + col%16, K = a_kk(h,e)
__global__ void conv_w_swz(const float* __restrict__ cheb, __bf16* __restrict__ B) {
    int idx = blockIdx.x * 256 + threadIdx.x;          // over 21*44*512
    if (idx >= KT_CHEB * NT_CHEB * 512) return;
    int t = idx >> 9, lane = (idx >> 4) & 31, e = idx & 15;
    int ktile = t / NT_CHEB, ntile = t % NT_CHEB;
    int h = lane >> 4, lr = lane & 15;
    int k = ktile * 32 + a_kk(h, e);
    int n = ntile * 16 + lr;
    float v = 0.f;
    if (k < N_FEAT && n < N_FEAT)
        v = cheb[(size_t)k * N_FEAT + n]
          - cheb[2 * (size_t)N_FEAT * N_FEAT + (size_t)k * N_FEAT + n];
    B[idx] = (__bf16)v;
}
__global__ void conv_fc1w_swz(const float* __restrict__ w, __bf16* __restrict__ o) {
    int idx = blockIdx.x * 256 + threadIdx.x;          // over 1314*16*512
    if (idx >= KT_FC1 * 16 * 512) return;
    int t = idx >> 9, lane = (idx >> 4) & 31, e = idx & 15;
    int ktile = t >> 4, ntile = t & 15;
    int h = lane >> 4, lr = lane & 15;
    int k = ktile * 32 + a_kk(h, e);
    int n = ntile * 16 + lr;
    o[idx] = (__bf16)w[(size_t)k * 256 + n];
}
__global__ void zero_f(float* __restrict__ p, int n) {
    int idx = blockIdx.x * 256 + threadIdx.x;
    if (idx < n) p[idx] = 0.f;
}

// --------------------------------------------------------- K4: cheb WMMA GEMM
// All operands pre-swizzled; each wave owns a 32x32 output tile, streams K
// straight from L2 with b128 loads, no LDS. Epilogue fuses +bias, +cor,
// softplus, bn1 and stores bf16 directly in fc1's A-swizzled layout.
__global__ __launch_bounds__(256) void cheb_gemm(
    const __bf16* __restrict__ A, const __bf16* __restrict__ B,
    const float* __restrict__ cor, const float* __restrict__ bias,
    const float* __restrict__ bn1, __bf16* __restrict__ act) {

    const int wave = (blockIdx.x * 256 + threadIdx.x) >> 5;
    const int lane = threadIdx.x & 31;
    const int half = lane >> 4, lr = lane & 15;
    const int tM = wave / (NPAD / 32);      // 0..511
    const int tN = wave % (NPAD / 32);      // 0..21
    const int mt0 = tM * 2, nt0 = tN * 2;

    v8f acc[2][2];
    acc[0][0] = acc[0][1] = acc[1][0] = acc[1][1] = (v8f){};

    const __bf16* ap0 = A + ((size_t)(mt0)     * KT_CHEB) * 512 + lane * 16;
    const __bf16* ap1 = A + ((size_t)(mt0 + 1) * KT_CHEB) * 512 + lane * 16;

    for (int kt = 0; kt < KT_CHEB; ++kt) {
        const __bf16* bp = B + ((size_t)kt * NT_CHEB + nt0) * 512 + lane * 16;
        if (kt + 2 < KT_CHEB) {
            __builtin_prefetch(ap0 + (kt + 2) * 512, 0, 1);
            __builtin_prefetch(ap1 + (kt + 2) * 512, 0, 1);
        }
        v16bf a0 = *(const v16bf*)(ap0 + (size_t)kt * 512);
        v16bf a1 = *(const v16bf*)(ap1 + (size_t)kt * 512);
        v16bf b0 = *(const v16bf*)(bp);
        v16bf b1 = *(const v16bf*)(bp + 512);
        acc[0][0] = wmma_bf16(a0, b0, acc[0][0]);
        acc[0][1] = wmma_bf16(a0, b1, acc[0][1]);
        acc[1][0] = wmma_bf16(a1, b0, acc[1][0]);
        acc[1][1] = wmma_bf16(a1, b1, acc[1][1]);
    }

    #pragma unroll
    for (int i = 0; i < 2; ++i)
        #pragma unroll
        for (int j = 0; j < 2; ++j) {
            int col = (nt0 + j) * 16 + lr;
            if (col < N_FEAT) {
                float g  = bn1[col], bb = bn1[N_FEAT + col];
                float mu = bn1[2 * N_FEAT + col];
                float iv = rsqrtf(bn1[3 * N_FEAT + col] + EPS);
                float bv = bias[col];
                #pragma unroll
                for (int v = 0; v < 8; ++v) {
                    int row = (mt0 + i) * 16 + half * 8 + v;   // 0..16383
                    float xv = acc[i][j][v] + bv;
                    if (row < N_NODES) xv += cor[row * N_FEAT + col];
                    float o = g * (softplusf(xv) - mu) * iv + bb;
                    // store in fc1 A-swizzled layout
                    int r  = row >> 6;                         // fc1 row 0..255
                    int kk = (row & 63) * N_FEAT + col;        // fc1 K 0..42047
                    int mt = r >> 4, lrm = r & 15;
                    int akt = kk >> 5, kkm = kk & 31;
                    int h2 = (kkm < 16) ? (kkm >> 3) : ((kkm - 16) >> 3);
                    int e2 = (kkm < 16) ? (kkm & 7) : (8 + (kkm & 7));
                    size_t oidx = ((size_t)mt * KT_FC1 + akt) * 512
                                + (h2 * 16 + lrm) * 16 + e2;
                    act[oidx] = (__bf16)o;
                }
            }
        }
}

// ------------------------------------------------ K5: fc1 WMMA GEMM (split-K)
// Wave owns one 16x32 tile for one K chunk; operands pre-swizzled, no LDS.
__global__ __launch_bounds__(256) void fc1_gemm(
    const __bf16* __restrict__ A, const __bf16* __restrict__ B,
    float* __restrict__ cbuf) {

    const int wave = (blockIdx.x * 256 + threadIdx.x) >> 5;   // 0..4095
    const int lane = threadIdx.x & 31;
    const int half = lane >> 4, lr = lane & 15;
    const int mt  = wave >> 8;                                // 0..15
    const int ntp = (wave >> 5) & 7;                          // 0..7
    const int kc  = wave & 31;                                // 0..31
    const int nt0 = ntp * 2;

    v8f acc[2];
    acc[0] = acc[1] = (v8f){};

    const int ktbeg = kc * 42;
    const int ktend = min(KT_FC1, ktbeg + 42);
    const __bf16* ap = A + ((size_t)mt * KT_FC1) * 512 + lane * 16;

    for (int kt = ktbeg; kt < ktend; ++kt) {
        const __bf16* bp = B + ((size_t)kt * 16 + nt0) * 512 + lane * 16;
        if (kt + 2 < ktend) __builtin_prefetch(ap + (size_t)(kt + 2) * 512, 0, 1);
        v16bf a  = *(const v16bf*)(ap + (size_t)kt * 512);
        v16bf b0 = *(const v16bf*)(bp);
        v16bf b1 = *(const v16bf*)(bp + 512);
        acc[0] = wmma_bf16(a, b0, acc[0]);
        acc[1] = wmma_bf16(a, b1, acc[1]);
    }

    #pragma unroll
    for (int j = 0; j < 2; ++j) {
        int col = (nt0 + j) * 16 + lr;
        #pragma unroll
        for (int v = 0; v < 8; ++v) {
            int row = mt * 16 + half * 8 + v;
            atomicAdd(&cbuf[row * 256 + col], acc[j][v]);
        }
    }
}

// -------------------------------------------------------------- fc1 epilogue
__global__ void fc1_epi(const float* __restrict__ cbuf, const float* __restrict__ b,
                        const float* __restrict__ bnf1, float* __restrict__ h) {
    int idx = blockIdx.x * 256 + threadIdx.x;
    if (idx >= 256 * 256) return;
    int col = idx & 255;
    h[idx] = bn_apply(bnf1, 256, col, softplusf(cbuf[idx] + b[col]));
}

// ----------------------------------------- head: fc2 -> fc3 -> softmax (fp32)
__global__ __launch_bounds__(32) void head_k(
    const float* __restrict__ h, const float* __restrict__ w2, const float* __restrict__ b2,
    const float* __restrict__ bn2, const float* __restrict__ w3, const float* __restrict__ b3,
    const float* __restrict__ bn3, float* __restrict__ out) {
    __shared__ float y2[32];
    __shared__ float y3[4];
    const int r = blockIdx.x, t = threadIdx.x;
    float s = b2[t];
    for (int k = 0; k < 256; ++k) s += h[r * 256 + k] * w2[k * 32 + t];
    y2[t] = bn_apply(bn2, 32, t, softplusf(s));
    __syncthreads();
    if (t < 4) {
        float s3 = b3[t];
        for (int k = 0; k < 32; ++k) s3 += y2[k] * w3[k * 4 + t];
        y3[t] = softplusf(bn_apply(bn3, 4, t, s3));    // bn THEN softplus here
    }
    __syncthreads();
    if (t < 4) {
        float m = fmaxf(fmaxf(y3[0], y3[1]), fmaxf(y3[2], y3[3]));
        float den = expf(y3[0] - m) + expf(y3[1] - m) + expf(y3[2] - m) + expf(y3[3] - m);
        out[r * 4 + t] = expf(y3[t] - m) / den;
    }
}

// ---------------------------------------------------------------------------
extern "C" void kernel_launch(void* const* d_in, const int* in_sizes, int n_in,
                              void* d_out, int out_size, void* d_ws, size_t ws_size,
                              hipStream_t stream) {
    (void)in_sizes; (void)n_in; (void)out_size; (void)ws_size;
    const float* x     = (const float*)d_in[0];
    const float* adj   = (const float*)d_in[1];
    const int*   ei    = (const int*)d_in[2];
    const float* chebw = (const float*)d_in[3];
    const float* chebb = (const float*)d_in[4];
    const float* fc1w  = (const float*)d_in[5];
    const float* fc1b  = (const float*)d_in[6];
    const float* fc2w  = (const float*)d_in[7];
    const float* fc2b  = (const float*)d_in[8];
    const float* fc3w  = (const float*)d_in[9];
    const float* fc3b  = (const float*)d_in[10];
    const float* bn1   = (const float*)d_in[11];
    const float* bnf1  = (const float*)d_in[12];
    const float* bnf2  = (const float*)d_in[13];
    const float* bnf3  = (const float*)d_in[14];
    float* out = (float*)d_out;

    char* p = (char*)d_ws;
    auto alloc = [&](size_t bytes) -> char* {
        char* r = p; p += (bytes + 255) & ~(size_t)255; return r;
    };
    float*  S    = (float*)alloc(4096 * 4);
    float*  h1   = (float*)alloc((size_t)FIN * 4);
    float*  h2   = (float*)alloc((size_t)FIN * 4);
    float*  cor  = (float*)alloc((size_t)FIN * 4);
    float*  cbuf = (float*)alloc(65536 * 4);
    float*  hfc1 = (float*)alloc(65536 * 4);
    __bf16* Abf  = (__bf16*)alloc((size_t)(M_TOT / 16) * KT_CHEB * 512 * 2);
    __bf16* Bbf  = (__bf16*)alloc((size_t)KT_CHEB * NT_CHEB * 512 * 2);
    __bf16* act  = (__bf16*)alloc((size_t)16 * KT_FC1 * 512 * 2);
    __bf16* w1b  = (__bf16*)alloc((size_t)KT_FC1 * 16 * 512 * 2);

    build_S<<<1, 512, 0, stream>>>(adj, ei, S);
    conv_x_swz<<<((M_TOT / 16) * KT_CHEB * 512 + 255) / 256, 256, 0, stream>>>(x, Abf);
    conv_w_swz<<<(KT_CHEB * NT_CHEB * 512 + 255) / 256, 256, 0, stream>>>(chebw, Bbf);
    conv_fc1w_swz<<<(KT_FC1 * 16 * 512 + 255) / 256, 256, 0, stream>>>(fc1w, w1b);
    zero_f<<<256, 256, 0, stream>>>(cbuf, 65536);
    spmm64<<<(FIN + 255) / 256, 256, 0, stream>>>(S, x, h1, 1.0f);
    spmm64<<<(FIN + 255) / 256, 256, 0, stream>>>(S, h1, h2, 2.0f);
    cor_kernel<<<(FIN + 255) / 256, 256, 0, stream>>>(h1, h2, chebw, cor);

    // 512 * 22 waves, 8 waves/block
    cheb_gemm<<<1408, 256, 0, stream>>>(Abf, Bbf, cor, chebb, bn1, act);
    // 16 * 8 * 32 = 4096 waves, 8 waves/block
    fc1_gemm<<<512, 256, 0, stream>>>(act, w1b, cbuf);
    fc1_epi<<<256, 256, 0, stream>>>(cbuf, fc1b, bnf1, hfc1);
    head_k<<<256, 32, 0, stream>>>(hfc1, fc2w, fc2b, bnf2, fc3w, fc3b, bnf3, out);
}
